// MOE_25065429139931
// MI455X (gfx1250) — compile-verified, matrix operand order checked by reading, not run on hardware
//
#include <hip/hip_runtime.h>

// ---------------------------------------------------------------------------
// MoE soft-routing classifier for MI455X (gfx1250, wave32, WMMA bf16).
// Dominant GEMM: 8 x (32768x512)@(512x512)  -> v_wmma_f32_16x16x32_bf16.
// ---------------------------------------------------------------------------

#define N_ROWS 32768
#define HDIM   512
#define SRC    8
#define RDIM   64
#define TDIM   12

typedef __attribute__((ext_vector_type(16))) __bf16 bf16x16;
typedef __attribute__((ext_vector_type(8)))  float  v8f;

union Frag {
    bf16x16 f;
    uint4   u[2];   // 2 x 16B = 32B = 16 bf16
};

__device__ __forceinline__ unsigned short f2bf(float x) {
    unsigned int u = __float_as_uint(x);
    unsigned int r = (u + 0x7FFFu + ((u >> 16) & 1u)) >> 16;   // round-nearest-even
    return (unsigned short)r;
}

// ---------------------------- prep kernels ---------------------------------

// W1t[s][n][k] = bf16( W1[s][k][n] )   (dest linear index == idx)
__global__ __launch_bounds__(256) void prep_w1t(const float* __restrict__ W1,
                                                unsigned short* __restrict__ W1t) {
    int idx = blockIdx.x * 256 + threadIdx.x;          // 8*512*512
    int s   = idx >> 18;
    int rem = idx & 262143;
    int n   = rem >> 9;          // output column
    int k   = rem & 511;         // contraction (h)
    W1t[idx] = f2bf(W1[s * 262144 + k * 512 + n]);
}

// Ust[c][h] = bf16( Us[s][h][r] ),  c = s*64 + r
__global__ __launch_bounds__(256) void prep_ust(const float* __restrict__ Us,
                                                unsigned short* __restrict__ Ust) {
    int idx = blockIdx.x * 256 + threadIdx.x;          // 512*512
    int c = idx >> 9;
    int h = idx & 511;
    int s = c >> 6;
    int r = c & 63;
    Ust[idx] = f2bf(Us[s * 32768 + h * 64 + r]);
}

// W2t[s][t][k] = bf16( W2[s][k][t] ), t padded to 16 with zeros
__global__ __launch_bounds__(256) void prep_w2t(const float* __restrict__ W2,
                                                unsigned short* __restrict__ W2t) {
    int idx = blockIdx.x * 256 + threadIdx.x;          // 8*16*512
    int s = idx >> 13;
    int t = (idx >> 9) & 15;
    int k = idx & 511;
    W2t[idx] = (t < TDIM) ? f2bf(W2[s * (HDIM * TDIM) + k * TDIM + t]) : (unsigned short)0;
}

// muU[c] = sum_h domain_encs[s][h] * Us[s][h][r]
__global__ __launch_bounds__(256) void prep_muu(const float* __restrict__ dom,
                                                const float* __restrict__ Us,
                                                float* __restrict__ muU) {
    int c = blockIdx.x * 256 + threadIdx.x;            // 512
    int s = c >> 6;
    int r = c & 63;
    float acc = 0.f;
    for (int h = 0; h < HDIM; ++h)
        acc += dom[s * HDIM + h] * Us[s * 32768 + h * 64 + r];
    muU[c] = acc;
}

// ------------------------- routing (alphas) kernel -------------------------
// P = enc @ Ust^T  (512 cols = 8 sources x 64 ranks), dist = ||P - muU||,
// alphas = softmax_s(-dist).  32 rows per block, 8 waves, WMMA bf16.
__global__ __launch_bounds__(256) void route_kernel(const float* __restrict__ enc,
                                                    const unsigned short* __restrict__ Ust,
                                                    const float* __restrict__ muU,
                                                    float* __restrict__ alphas) {
    __shared__ unsigned short lAu[32 * 512];
    __shared__ float distAcc[32 * 8];

    const int tid  = threadIdx.x;
    const int row0 = blockIdx.x * 32;

    // stage 32x512 activation tile as bf16
    const float4* enc4 = (const float4*)(enc + (size_t)row0 * HDIM);
    for (int i = 0; i < 16; ++i) {
        int idx = tid + i * 256;
        float4 v = enc4[idx];
        int b = idx * 4;
        lAu[b + 0] = f2bf(v.x); lAu[b + 1] = f2bf(v.y);
        lAu[b + 2] = f2bf(v.z); lAu[b + 3] = f2bf(v.w);
    }
    distAcc[tid] = 0.f;                       // 256 == 32*8
    __syncthreads();

    const int wave = tid >> 5, lane = tid & 31;
    const int mBase = (wave & 1) * 16;
    const int nSub  = (wave >> 1) * 16;
    const int n15   = lane & 15;
    const int kh    = (lane >> 4) << 3;       // K half select (A/B frag)
    const int mOff  = (lane >> 4) << 3;       // C frag row offset

    const unsigned short* arow = lAu + (mBase + n15) * 512;
    for (int cb = 0; cb < 8; ++cb) {
        const int nBase = cb * 64 + nSub;
        const unsigned short* brow = Ust + (size_t)(nBase + n15) * 512;
        v8f acc = {};
        for (int kb = 0; kb < 512; kb += 32) {
            Frag a, b;
            a.u[0] = *(const uint4*)(arow + kb + kh);
            a.u[1] = *(const uint4*)(arow + kb + kh + 16);
            b.u[0] = *(const uint4*)(brow + kb + kh);
            b.u[1] = *(const uint4*)(brow + kb + kh + 16);
            acc = __builtin_amdgcn_wmma_f32_16x16x32_bf16(false, a.f, false, b.f,
                                                          (short)0, acc, false, false);
        }
        for (int v = 0; v < 8; ++v) {
            int m = mBase + v + mOff;
            int c = nBase + n15;
            float d = acc[v] - muU[c];
            atomicAdd(&distAcc[m * 8 + (c >> 6)], d * d);
        }
    }
    __syncthreads();

    if (tid < 32) {                           // one lane per row: sqrt + softmax over 8
        float d[SRC], e[SRC];
        float dmin = 3.4e38f;
        for (int s = 0; s < SRC; ++s) { d[s] = sqrtf(distAcc[tid * 8 + s]); dmin = fminf(dmin, d[s]); }
        float sum = 0.f;
        for (int s = 0; s < SRC; ++s) { e[s] = expf(dmin - d[s]); sum += e[s]; }
        float inv = 1.f / sum;
        for (int s = 0; s < SRC; ++s) alphas[(size_t)(row0 + tid) * 8 + s] = e[s] * inv;
    }
}

// ---------------------------- expert kernel --------------------------------
// For each source s: H1 = relu(relu(enc)@W1t[s] + b1), staged bf16 in LDS,
// OUT = sigmoid(H1@W2t[s] + b2); accumulate alpha[s]*OUT in registers.
__global__ __launch_bounds__(256) void expert_kernel(const float* __restrict__ enc,
                                                     const unsigned short* __restrict__ W1t,
                                                     const float* __restrict__ b1,
                                                     const unsigned short* __restrict__ W2t,
                                                     const float* __restrict__ b2,
                                                     const float* __restrict__ alphas,
                                                     float* __restrict__ out) {
    __shared__ unsigned short lAu[32 * 512];   // relu(enc) tile, bf16
    __shared__ unsigned short lH1[32 * 512];   // hidden tile, bf16

    const int tid  = threadIdx.x;
    const int row0 = blockIdx.x * 32;

    const float4* enc4 = (const float4*)(enc + (size_t)row0 * HDIM);
    for (int i = 0; i < 16; ++i) {
        int idx = tid + i * 256;
        float4 v = enc4[idx];
        int b = idx * 4;
        lAu[b + 0] = f2bf(fmaxf(v.x, 0.f)); lAu[b + 1] = f2bf(fmaxf(v.y, 0.f));
        lAu[b + 2] = f2bf(fmaxf(v.z, 0.f)); lAu[b + 3] = f2bf(fmaxf(v.w, 0.f));
    }
    __syncthreads();

    const int wave = tid >> 5, lane = tid & 31;
    const int mBase = (wave & 1) * 16;
    const int nSub  = (wave >> 1) * 16;
    const int n15   = lane & 15;
    const int kh    = (lane >> 4) << 3;
    const int mOff  = (lane >> 4) << 3;

    v8f outAcc = {};                          // waves 0,1: persistent weighted output

    for (int s = 0; s < SRC; ++s) {
        // ---- GEMM1: 32x512 hidden tile, all 8 waves ----
        const unsigned short* Wb   = W1t + (size_t)s * 262144;
        const unsigned short* arow = lAu + (mBase + n15) * 512;
        for (int cb = 0; cb < 8; ++cb) {
            const int nBase = cb * 64 + nSub;
            const unsigned short* brow = Wb + (size_t)(nBase + n15) * 512;
            v8f acc = {};
            for (int kb = 0; kb < 512; kb += 32) {
                Frag a, b;
                a.u[0] = *(const uint4*)(arow + kb + kh);
                a.u[1] = *(const uint4*)(arow + kb + kh + 16);
                b.u[0] = *(const uint4*)(brow + kb + kh);
                b.u[1] = *(const uint4*)(brow + kb + kh + 16);
                acc = __builtin_amdgcn_wmma_f32_16x16x32_bf16(false, a.f, false, b.f,
                                                              (short)0, acc, false, false);
            }
            for (int v = 0; v < 8; ++v) {
                int m = mBase + v + mOff;
                int c = nBase + n15;
                float h = fmaxf(acc[v] + b1[s * HDIM + c], 0.f);
                lH1[m * 512 + c] = f2bf(h);
            }
        }
        __syncthreads();

        // ---- GEMM2: 32x16 (T padded) output tile, waves 0,1 ----
        if (wave < 2) {
            const int mB2 = wave * 16;
            const unsigned short* arow2 = lH1 + (mB2 + n15) * 512;
            const unsigned short* brow2 = W2t + (size_t)s * 8192 + (size_t)n15 * 512;
            v8f acc = {};
            for (int kb = 0; kb < 512; kb += 32) {
                Frag a, b;
                a.u[0] = *(const uint4*)(arow2 + kb + kh);
                a.u[1] = *(const uint4*)(arow2 + kb + kh + 16);
                b.u[0] = *(const uint4*)(brow2 + kb + kh);
                b.u[1] = *(const uint4*)(brow2 + kb + kh + 16);
                acc = __builtin_amdgcn_wmma_f32_16x16x32_bf16(false, a.f, false, b.f,
                                                              (short)0, acc, false, false);
            }
            if (n15 < TDIM) {
                for (int v = 0; v < 8; ++v) {
                    int m = mB2 + v + mOff;
                    float o = acc[v] + b2[s * TDIM + n15];
                    o = 1.f / (1.f + expf(-o));
                    float al = alphas[(size_t)(row0 + m) * 8 + s];
                    outAcc[v] += al * o;
                }
            }
        }
        __syncthreads();      // protect lH1 before next source overwrites it
    }

    if (wave < 2 && n15 < TDIM) {
        const int mB2 = wave * 16;
        for (int v = 0; v < 8; ++v) {
            int m = mB2 + v + mOff;
            out[(size_t)(row0 + m) * TDIM + n15] = outAcc[v];
        }
    }
}

// ------------------------------- launcher ----------------------------------

extern "C" void kernel_launch(void* const* d_in, const int* in_sizes, int n_in,
                              void* d_out, int out_size, void* d_ws, size_t ws_size,
                              hipStream_t stream) {
    (void)in_sizes; (void)n_in; (void)out_size; (void)ws_size;

    const float* enc = (const float*)d_in[0];   // [N,H]
    const float* dom = (const float*)d_in[1];   // [S,H]
    const float* Us  = (const float*)d_in[2];   // [S,H,R]
    const float* W1  = (const float*)d_in[3];   // [S,H,H]
    const float* b1  = (const float*)d_in[4];   // [S,H]
    const float* W2  = (const float*)d_in[5];   // [S,H,T]
    const float* b2  = (const float*)d_in[6];   // [S,T]
    float* out = (float*)d_out;                 // [N,T]

    // workspace carve-out (~5.9 MB)
    char* ws = (char*)d_ws;
    unsigned short* W1t = (unsigned short*)(ws);                       // 8*512*512*2 = 4 MB
    unsigned short* Ust = (unsigned short*)(ws + 4194304);             // 512*512*2
    unsigned short* W2t = (unsigned short*)(ws + 4194304 + 524288);    // 8*16*512*2
    float* muU    = (float*)(ws + 4194304 + 524288 + 131072);          // 512*4
    float* alphas = (float*)(ws + 4194304 + 524288 + 131072 + 2048);   // 32768*8*4

    prep_w1t<<<8192, 256, 0, stream>>>(W1, W1t);
    prep_ust<<<1024, 256, 0, stream>>>(Us, Ust);
    prep_w2t<<<256,  256, 0, stream>>>(W2, W2t);
    prep_muu<<<2,    256, 0, stream>>>(dom, Us, muU);

    route_kernel<<<N_ROWS / 32, 256, 0, stream>>>(enc, Ust, muU, alphas);
    expert_kernel<<<N_ROWS / 32, 256, 0, stream>>>(enc, W1t, b1, W2t, b2, alphas, out);
}
